// PiecewiseDiscontinuousPolynomial_19164144075552
// MI455X (gfx1250) — matrix operand order repaired and checked by reading, not compiled, not run on hardware
//
#include <hip/hip_runtime.h>
#include <hip/hip_bf16.h>

// ---- CDNA5 vector types (ext vectors: trivially copyable, union-safe) ----
typedef __attribute__((ext_vector_type(16))) _Float16 v16h;
typedef __attribute__((ext_vector_type(8)))  float    v8f;
typedef __attribute__((ext_vector_type(4)))  unsigned u32x4;
typedef __attribute__((ext_vector_type(2)))  unsigned u32x2;
typedef __attribute__((ext_vector_type(4)))  float    f32x4;

union V16 { v16h v; u32x4 q[2]; };
union HPK { _Float16 h[4]; u32x2 u; };

#define IN_F   128
#define OUT_F  128
#define BATCHN 1024
#define KDIM   128   // N * SEGMENTS
#define TB     32    // batch rows per workgroup (2 M-tiles sharing each B frag)

// ---------------------------------------------------------------------------
// Kernel 1: convert w (f32, [o][i][k]) -> f16 in workspace, same layout.
// ---------------------------------------------------------------------------
__global__ __launch_bounds__(256) void convert_w_kernel(
    const float* __restrict__ w, _Float16* __restrict__ wh, int n) {
  int idx = (blockIdx.x * 256 + threadIdx.x) * 4;
  if (idx + 3 < n) {
    f32x4 v = *reinterpret_cast<const f32x4*>(w + idx);
    HPK p;
    p.h[0] = (_Float16)v.x; p.h[1] = (_Float16)v.y;
    p.h[2] = (_Float16)v.z; p.h[3] = (_Float16)v.w;
    *reinterpret_cast<u32x2*>(wh + idx) = p.u;
  }
}

// ---------------------------------------------------------------------------
// Kernel 2: block = 64 threads (2 wave32). WG = 32 batch rows x 32 outputs;
// each wave owns one 16-wide o-tile and both 16-row M-tiles (B frag reused by
// two v_wmma per chunk -> halves L2 gather traffic). Per input feature i:
//   F_i (32x16) = A_i (32x128 densified sparse basis) x W_i (128x16)
// as 2 M-tiles x 4 chained v_wmma_f32_16x16x32_f16; then sum/prod accumulate.
// Double-buffered A tile: one barrier per feature.
// ---------------------------------------------------------------------------
__global__ __launch_bounds__(64) void pdp_kernel(
    const float* __restrict__ x, const _Float16* __restrict__ wh,
    const float* __restrict__ w_sum, const float* __restrict__ w_prod,
    float* __restrict__ out) {
  __shared__ u32x2         basisPk[IN_F * TB];                 // 32 KB
  __shared__ unsigned char idArr[IN_F * TB];                   //  4 KB
  __shared__ __align__(16) _Float16 Atile[2][TB * KDIM];       // 2 x 8 KB

  const int tid   = threadIdx.x;
  const int bbase = blockIdx.x * TB;
  const int lane  = tid & 31;
  const int wave  = tid >> 5;
  const int obase = blockIdx.y * 32 + wave * 16;

  // ---- Phase 0: segment id + Lagrange basis (nodes [-1,-0.5,0.5,1]) ----
  for (int t = 0; t < (IN_F * TB) / 64; ++t) {
    int p   = tid + t * 64;
    int row = p & (TB - 1);
    int i   = p >> 5;
    float xv = x[(size_t)(bbase + row) * IN_F + i];
    int id = (int)((xv + 1.0f) * 16.0f);
    id = id < 0 ? 0 : (id > 31 ? 31 : id);
    float u  = 32.0f * xv - 2.0f * (float)id + 31.0f;   // x_in in [-1,1]
    float fa = u + 1.0f, fb = u + 0.5f, fc = u - 0.5f, fd = u - 1.0f;
    float L0 = fb * fc * fd * (-1.0f / 1.5f);
    float L1 = fa * fc * fd * ( 1.0f / 0.75f);
    float L2 = fa * fb * fd * (-1.0f / 0.75f);
    float L3 = fa * fb * fc * ( 1.0f / 1.5f);
    HPK pk;
    pk.h[0] = (_Float16)L0; pk.h[1] = (_Float16)L1;
    pk.h[2] = (_Float16)L2; pk.h[3] = (_Float16)L3;
    basisPk[p] = pk.u;
    idArr[p]   = (unsigned char)id;
  }

  const int lrow  = lane & 15;   // N (output col) / A row within tile
  const int lhalf = lane >> 4;   // lane half -> K sub-block
  const _Float16* wptr = wh + (size_t)(obase + lrow) * (IN_F * KDIM);

  // A-tile build assignment: 2 threads per row, one 128B half-row each.
  const int brow  = tid >> 1;    // 0..31
  const int bhalf = tid & 1;     // which 16-slot half of the 256B row

  // Build = zero half-row (8x ds_store_b128) + one conditional ds_store_b64
  // of the 4xf16 basis pack at k = 4*id (same-wave LDS ops are in-order).
  auto build = [&](int i, int buf) {
    unsigned id = (unsigned)idArr[i * TB + brow];
    u32x2 pk    = basisPk[i * TB + brow];
    _Float16* base = &Atile[buf][0];
    u32x4* dst = reinterpret_cast<u32x4*>(base) + brow * 16 + bhalf * 8;
    u32x4 z = {0u, 0u, 0u, 0u};
#pragma unroll
    for (int q = 0; q < 8; ++q) dst[q] = z;
    if ((id >> 4) == (unsigned)bhalf) {
      *reinterpret_cast<u32x2*>(base + brow * KDIM + id * 4) = pk;
    }
  };

  v8f accS0 = {0.f,0.f,0.f,0.f,0.f,0.f,0.f,0.f};
  v8f accP0 = {1.f,1.f,1.f,1.f,1.f,1.f,1.f,1.f};
  v8f accS1 = {0.f,0.f,0.f,0.f,0.f,0.f,0.f,0.f};
  v8f accP1 = {1.f,1.f,1.f,1.f,1.f,1.f,1.f,1.f};

  __syncthreads();           // basis tables ready
  build(0, 0);
  __syncthreads();           // A(0) ready

  for (int i = 0; i < IN_F; ++i) {
    const int buf = i & 1;
    const _Float16* At  = &Atile[buf][0];
    const _Float16* bsl = wptr + (size_t)i * KDIM;

    v8f d0 = {0.f,0.f,0.f,0.f,0.f,0.f,0.f,0.f};
    v8f d1 = {0.f,0.f,0.f,0.f,0.f,0.f,0.f,0.f};
#pragma unroll
    for (int c = 0; c < 4; ++c) {
      V16 a0, a1, bf;
      const u32x4* ap0 = reinterpret_cast<const u32x4*>(
          At + lrow * KDIM + c * 32 + lhalf * 8);
      const u32x4* ap1 = reinterpret_cast<const u32x4*>(
          At + (16 + lrow) * KDIM + c * 32 + lhalf * 8);
      a0.q[0] = ap0[0];  a0.q[1] = ap0[2];   // +32B
      a1.q[0] = ap1[0];  a1.q[1] = ap1[2];
      const u32x4* bp = reinterpret_cast<const u32x4*>(bsl + c * 32 + lhalf * 8);
      bf.q[0] = bp[0];   bf.q[1] = bp[2];
      d0 = __builtin_amdgcn_wmma_f32_16x16x32_f16(
          false, a0.v, false, bf.v, (short)0, d0, false, false);
      d1 = __builtin_amdgcn_wmma_f32_16x16x32_f16(
          false, a1.v, false, bf.v, (short)0, d1, false, false);
    }
    accS0 += d0;  accP0 *= d0;
    accS1 += d1;  accP1 *= d1;

    if (i + 1 < IN_F) {
      __builtin_prefetch(bsl + KDIM, 0, 1);   // next-i B slice
      build(i + 1, buf ^ 1);                  // build into other buffer
    }
    __syncthreads();                          // single barrier per feature
  }

  // ---- Epilogue: D layout — VGPR r -> M = r + 8*lhalf, N = lrow ----
  const int o    = obase + lrow;
  const float wsv = w_sum[o];
  const float wpv = w_prod[o];
  const int b0 = bbase + lhalf * 8;
#pragma unroll
  for (int r = 0; r < 8; ++r) {
    out[(size_t)(b0 + r) * OUT_F + o]      = accS0[r] * wsv + accP0[r] * wpv;
  }
#pragma unroll
  for (int r = 0; r < 8; ++r) {
    out[(size_t)(b0 + 16 + r) * OUT_F + o] = accS1[r] * wsv + accP1[r] * wpv;
  }
}

// ---------------------------------------------------------------------------
extern "C" void kernel_launch(void* const* d_in, const int* in_sizes, int n_in,
                              void* d_out, int out_size, void* d_ws, size_t ws_size,
                              hipStream_t stream) {
  const float* x      = (const float*)d_in[0];
  const float* w      = (const float*)d_in[1];
  const float* w_sum  = (const float*)d_in[2];
  const float* w_prod = (const float*)d_in[3];
  float* out = (float*)d_out;
  _Float16* wh = (_Float16*)d_ws;   // 4 MB f16 copy of w

  const int n = OUT_F * IN_F * KDIM;  // 2,097,152
  convert_w_kernel<<<dim3(n / 4 / 256), dim3(256), 0, stream>>>(w, wh, n);

  dim3 grid(BATCHN / TB, OUT_F / 32);  // (32, 4) = 128 workgroups, 2 waves each
  pdp_kernel<<<grid, dim3(64), 0, stream>>>(x, wh, w_sum, w_prod, out);
}
